// FeatureMatching_63763084476698
// MI455X (gfx1250) — compile-verified, matrix operand order checked by reading, not run on hardware
//
#include <hip/hip_runtime.h>
#include <hip/hip_bf16.h>
#include <math.h>

// ---------------------------------------------------------------------------
// FeatureMatching (entropic OT / Sinkhorn flow init) for MI455X (gfx1250).
//   B=2, C=256, N=4096.
// Phases:
//   1) prep: normalize feats -> f16 [B,N,C]; pack (x,y,z,|p|^2) float4s; init a.
//   2) gemm_exp: corr = f1 f2^T via v_wmma_f32_16x16x32_f16, fused distance
//      gate + exp -> K stored f16 [B,N,N] (64 MB total, L2-resident).
//   3) 10x { col_pass: b = (p2/(K^T a))^pow ; row_pass: a = (p1/(K b))^pow }
//   4) flow: T = a K b fused with row-sum and T*p2 -> flow [B,N,3].
// ---------------------------------------------------------------------------

typedef _Float16 v16h __attribute__((ext_vector_type(16)));
typedef _Float16 v8h  __attribute__((ext_vector_type(8)));
typedef float    v8f  __attribute__((ext_vector_type(8)));

#define BATCH 2
#define NPTS  4096
#define CDIM  256
#define TILE  128
#define KSTEP 32
#define LDK   (KSTEP + 8)   // padded LDS row stride (halves) -> conflict-free b128

// workspace layout (bytes)
static constexpr size_t SZ_FH   = (size_t)BATCH * NPTS * CDIM * 2;  // 4 MB each
static constexpr size_t SZ_PT   = (size_t)BATCH * NPTS * 16;
static constexpr size_t SZ_VEC  = (size_t)BATCH * NPTS * 4;
static constexpr size_t OFF_F1H = 0;
static constexpr size_t OFF_F2H = OFF_F1H + SZ_FH;
static constexpr size_t OFF_PT1 = OFF_F2H + SZ_FH;
static constexpr size_t OFF_PT2 = OFF_PT1 + SZ_PT;
static constexpr size_t OFF_A   = OFF_PT2 + SZ_PT;
static constexpr size_t OFF_B   = OFF_A + SZ_VEC;
static constexpr size_t OFF_K   = (OFF_B + SZ_VEC + 255) & ~(size_t)255;
// SZ_K = 2*4096*4096*2 = 64 MB (fits in 192 MB L2)

// ---------------------------------------------------------------------------
// 1) prep: feature L2-normalization + point packing + init of a-vector
// ---------------------------------------------------------------------------
__global__ __launch_bounds__(256) void prep_kernel(
    const float* __restrict__ pc, const float* __restrict__ ft,
    _Float16* __restrict__ fh, float4* __restrict__ pt,
    float* __restrict__ avec, int initA) {
  const int b = blockIdx.y;
  const int n = blockIdx.x * 256 + threadIdx.x;
  const size_t fbase = ((size_t)b * CDIM) * NPTS + n;

  float s = 0.f;
  for (int c = 0; c < CDIM; ++c) {
    float v = ft[fbase + (size_t)c * NPTS];
    s += v * v;
  }
  const float inv = rsqrtf(s + 1e-6f);
  _Float16* dst = fh + ((size_t)b * NPTS + n) * CDIM;
  for (int c = 0; c < CDIM; ++c) {
    float v = ft[fbase + (size_t)c * NPTS];
    dst[c] = (_Float16)(v * inv);
  }

  const float x = pc[((size_t)b * 3 + 0) * NPTS + n];
  const float y = pc[((size_t)b * 3 + 1) * NPTS + n];
  const float z = pc[((size_t)b * 3 + 2) * NPTS + n];
  pt[(size_t)b * NPTS + n] = make_float4(x, y, z, x * x + y * y + z * z);
  if (initA) avec[(size_t)b * NPTS + n] = 1.0f / NPTS;
}

// ---------------------------------------------------------------------------
// 2) WMMA GEMM (corr) fused with distance gate + exp -> K (f16)
//    128x128 tile per 256-thread block; 8 waves in 4x2, each 32x64 (2x4 accs).
// ---------------------------------------------------------------------------
__global__ __launch_bounds__(256) void gemm_exp_kernel(
    const _Float16* __restrict__ f1h, const _Float16* __restrict__ f2h,
    const float4* __restrict__ pt1, const float4* __restrict__ pt2,
    _Float16* __restrict__ Kh, const float* __restrict__ epsilon) {
  __shared__ _Float16 ash[TILE * LDK];
  __shared__ _Float16 bsh[TILE * LDK];

  const int b    = blockIdx.z;
  const int m0   = blockIdx.y * TILE;
  const int n0   = blockIdx.x * TILE;
  const int tid  = threadIdx.x;
  const int lane = tid & 31;
  const int wave = tid >> 5;
  const int waveM = wave & 3;   // 4 row-groups of 32
  const int waveN = wave >> 2;  // 2 col-groups of 64
  const int rowIn = lane & 15;
  const int hi    = lane >> 4;
  const int koff  = hi * 8;     // lanes 16-31 start at K=8 (and K=24)

  const _Float16* Abase = f1h + ((size_t)b * NPTS + m0) * CDIM;
  const _Float16* Bbase = f2h + ((size_t)b * NPTS + n0) * CDIM;

  v8f acc[2][4] = {};

  for (int kk = 0; kk < CDIM; kk += KSTEP) {
    // cooperative tile load: 512 x 16B units per tile, 2 units per thread
#pragma unroll
    for (int u = 0; u < 2; ++u) {
      const int id  = tid + u * 256;
      const int row = id >> 2;
      const int ch  = id & 3;
      *(uint4*)&ash[row * LDK + ch * 8] =
          *(const uint4*)&Abase[(size_t)row * CDIM + kk + ch * 8];
      *(uint4*)&bsh[row * LDK + ch * 8] =
          *(const uint4*)&Bbase[(size_t)row * CDIM + kk + ch * 8];
    }
    __syncthreads();

    v16h fa[2], fb[4];
#pragma unroll
    for (int i = 0; i < 2; ++i) {
      const int r = waveM * 32 + i * 16 + rowIn;
      const _Float16* p = &ash[r * LDK + koff];
      v8h lo  = *(const v8h*)p;         // K = koff .. koff+7
      v8h hi8 = *(const v8h*)(p + 16);  // K = koff+16 .. koff+23
      fa[i] = __builtin_shufflevector(lo, hi8, 0, 1, 2, 3, 4, 5, 6, 7,
                                      8, 9, 10, 11, 12, 13, 14, 15);
    }
#pragma unroll
    for (int j = 0; j < 4; ++j) {
      const int r = waveN * 64 + j * 16 + rowIn;
      const _Float16* p = &bsh[r * LDK + koff];
      v8h lo  = *(const v8h*)p;
      v8h hi8 = *(const v8h*)(p + 16);
      fb[j] = __builtin_shufflevector(lo, hi8, 0, 1, 2, 3, 4, 5, 6, 7,
                                      8, 9, 10, 11, 12, 13, 14, 15);
    }
#pragma unroll
    for (int i = 0; i < 2; ++i)
#pragma unroll
      for (int j = 0; j < 4; ++j)
        acc[i][j] = __builtin_amdgcn_wmma_f32_16x16x32_f16(
            false, fa[i], false, fb[j], (short)0, acc[i][j], false, false);
    __syncthreads();
  }

  const float eps    = __expf(epsilon[0]) + 0.03f;
  const float invEps = 1.0f / eps;

  // epilogue: per C/D layout, N = lane&15 (constant over i,r); M = r + 8*hi
  float4 q2[4];
#pragma unroll
  for (int j = 0; j < 4; ++j) {
    const int n = n0 + waveN * 64 + j * 16 + (lane & 15);
    q2[j] = pt2[(size_t)b * NPTS + n];
  }
#pragma unroll
  for (int i = 0; i < 2; ++i) {
#pragma unroll
    for (int r = 0; r < 8; ++r) {
      const int m = m0 + waveM * 32 + i * 16 + r + hi * 8;
      const float4 q1 = pt1[(size_t)b * NPTS + m];
#pragma unroll
      for (int j = 0; j < 4; ++j) {
        const int n = n0 + waveN * 64 + j * 16 + (lane & 15);
        const float corr = acc[i][j][r];
        const float dist = q1.w + q2[j].w -
            2.0f * (q1.x * q2[j].x + q1.y * q2[j].y + q1.z * q2[j].z);
        const float kv = (dist < 0.25f) ? __expf((corr - 1.0f) * invEps) : 0.0f;
        Kh[((size_t)b * NPTS + m) * NPTS + n] = (_Float16)kv;
      }
    }
  }
}

// ---------------------------------------------------------------------------
// 3a) column pass: KTa[m] = sum_n K[n,m] a[n]; b = (p2/(KTa+1e-8))^power
//     thread = one column; a preloaded to LDS; row reads are 512B coalesced.
// ---------------------------------------------------------------------------
__global__ __launch_bounds__(256) void col_pass_kernel(
    const _Float16* __restrict__ Kh, const float* __restrict__ avec,
    float* __restrict__ bvec, const float* __restrict__ gamma,
    const float* __restrict__ epsilon) {
  __shared__ float ash[NPTS];
  const int b = blockIdx.y;
  const int m = blockIdx.x * 256 + threadIdx.x;
  for (int i = threadIdx.x; i < NPTS; i += 256)
    ash[i] = avec[(size_t)b * NPTS + i];
  __syncthreads();

  const _Float16* Kb = Kh + (size_t)b * NPTS * NPTS;
  float acc = 0.f;
  for (int n = 0; n < NPTS; ++n)
    acc += (float)Kb[(size_t)n * NPTS + m] * ash[n];

  const float eps   = __expf(epsilon[0]) + 0.03f;
  const float gam   = __expf(gamma[0]) + 0.03f;
  const float power = gam / (gam + eps);
  bvec[(size_t)b * NPTS + m] = __powf((1.0f / NPTS) / (acc + 1e-8f), power);
}

// ---------------------------------------------------------------------------
// 3b) row pass: Kb[n] = sum_m K[n,m] b[m]; a = (p1/(Kb+1e-8))^power
//     wave-per-row (8 rows/wave), b128 lane loads, shuffle reduction.
// ---------------------------------------------------------------------------
__global__ __launch_bounds__(256) void row_pass_kernel(
    const _Float16* __restrict__ Kh, const float* __restrict__ bvec,
    float* __restrict__ avec, const float* __restrict__ gamma,
    const float* __restrict__ epsilon) {
  __shared__ float bsh[NPTS];
  const int b = blockIdx.y;
  for (int i = threadIdx.x; i < NPTS; i += 256)
    bsh[i] = bvec[(size_t)b * NPTS + i];
  __syncthreads();

  const int lane = threadIdx.x & 31;
  const int wave = threadIdx.x >> 5;
  const _Float16* Kb = Kh + (size_t)b * NPTS * NPTS;
  const float eps   = __expf(epsilon[0]) + 0.03f;
  const float gam   = __expf(gamma[0]) + 0.03f;
  const float power = gam / (gam + eps);

  for (int rr = 0; rr < 8; ++rr) {
    const int n = blockIdx.x * 64 + wave * 8 + rr;
    const _Float16* row = Kb + (size_t)n * NPTS;
    float acc = 0.f;
    for (int m = lane * 8; m < NPTS; m += 256) {
      v8h v = *(const v8h*)&row[m];
#pragma unroll
      for (int k = 0; k < 8; ++k) acc += (float)v[k] * bsh[m + k];
    }
#pragma unroll
    for (int off = 16; off > 0; off >>= 1) acc += __shfl_xor(acc, off, 32);
    if (lane == 0)
      avec[(size_t)b * NPTS + n] = __powf((1.0f / NPTS) / (acc + 1e-8f), power);
  }
}

// ---------------------------------------------------------------------------
// 4) final: T = a K b fused with row-sum and T*p2; flow = T p2 / rowsum - p1
// ---------------------------------------------------------------------------
__global__ __launch_bounds__(256) void flow_kernel(
    const _Float16* __restrict__ Kh, const float* __restrict__ avec,
    const float* __restrict__ bvec, const float4* __restrict__ pt1,
    const float4* __restrict__ pt2, float* __restrict__ out) {
  __shared__ float  bsh[NPTS];
  __shared__ float4 psh[NPTS];
  const int b = blockIdx.y;
  for (int i = threadIdx.x; i < NPTS; i += 256) {
    bsh[i] = bvec[(size_t)b * NPTS + i];
    psh[i] = pt2[(size_t)b * NPTS + i];
  }
  __syncthreads();

  const int lane = threadIdx.x & 31;
  const int wave = threadIdx.x >> 5;
  const _Float16* Kb = Kh + (size_t)b * NPTS * NPTS;

  for (int rr = 0; rr < 8; ++rr) {
    const int n = blockIdx.x * 64 + wave * 8 + rr;
    const _Float16* row = Kb + (size_t)n * NPTS;
    float rs = 0.f, fx = 0.f, fy = 0.f, fz = 0.f;
    for (int m = lane * 8; m < NPTS; m += 256) {
      v8h v = *(const v8h*)&row[m];
#pragma unroll
      for (int k = 0; k < 8; ++k) {
        const float kb = (float)v[k] * bsh[m + k];
        const float4 q = psh[m + k];
        rs += kb; fx += kb * q.x; fy += kb * q.y; fz += kb * q.z;
      }
    }
#pragma unroll
    for (int off = 16; off > 0; off >>= 1) {
      rs += __shfl_xor(rs, off, 32);
      fx += __shfl_xor(fx, off, 32);
      fy += __shfl_xor(fy, off, 32);
      fz += __shfl_xor(fz, off, 32);
    }
    if (lane == 0) {
      const float an = avec[(size_t)b * NPTS + n];
      const float denom = an * rs + 1e-6f;
      const float4 q1 = pt1[(size_t)b * NPTS + n];
      float* o = out + ((size_t)b * NPTS + n) * 3;
      o[0] = an * fx / denom - q1.x;
      o[1] = an * fy / denom - q1.y;
      o[2] = an * fz / denom - q1.z;
    }
  }
}

// ---------------------------------------------------------------------------
extern "C" void kernel_launch(void* const* d_in, const int* in_sizes, int n_in,
                              void* d_out, int out_size, void* d_ws,
                              size_t ws_size, hipStream_t stream) {
  (void)in_sizes; (void)n_in; (void)out_size; (void)ws_size;
  const float* pc1     = (const float*)d_in[0];
  const float* pc2     = (const float*)d_in[1];
  const float* ft1     = (const float*)d_in[2];
  const float* ft2     = (const float*)d_in[3];
  const float* gamma   = (const float*)d_in[4];
  const float* epsilon = (const float*)d_in[5];

  char* ws = (char*)d_ws;
  _Float16* f1h  = (_Float16*)(ws + OFF_F1H);
  _Float16* f2h  = (_Float16*)(ws + OFF_F2H);
  float4*   pt1  = (float4*)(ws + OFF_PT1);
  float4*   pt2  = (float4*)(ws + OFF_PT2);
  float*    avec = (float*)(ws + OFF_A);
  float*    bvec = (float*)(ws + OFF_B);
  _Float16* Kh   = (_Float16*)(ws + OFF_K);

  dim3 gPrep(NPTS / 256, BATCH);
  prep_kernel<<<gPrep, 256, 0, stream>>>(pc1, ft1, f1h, pt1, avec, 1);
  prep_kernel<<<gPrep, 256, 0, stream>>>(pc2, ft2, f2h, pt2, avec, 0);

  dim3 gG(NPTS / TILE, NPTS / TILE, BATCH);
  gemm_exp_kernel<<<gG, 256, 0, stream>>>(f1h, f2h, pt1, pt2, Kh, epsilon);

  dim3 gCol(NPTS / 256, BATCH);
  dim3 gRow(NPTS / 64, BATCH);
  for (int it = 0; it < 10; ++it) {
    col_pass_kernel<<<gCol, 256, 0, stream>>>(Kh, avec, bvec, gamma, epsilon);
    row_pass_kernel<<<gRow, 256, 0, stream>>>(Kh, bvec, avec, gamma, epsilon);
  }

  flow_kernel<<<gRow, 256, 0, stream>>>(Kh, avec, bvec, pt1, pt2,
                                        (float*)d_out);
}
